// DispatcherLayer_987842478501
// MI455X (gfx1250) — compile-verified, matrix operand order checked by reading, not run on hardware
//
#include <hip/hip_runtime.h>
#include <hip/hip_bf16.h>

#define L_SEQ 4096
#define B_SZ  4
#define E_DIM 1024
#define H_NUM 16
#define HD    64
#define LVLS  12
#define MROWS (B_SZ * L_SEQ)   // 16384
#define KDIM  1024
#define NATT  (LVLS * H_NUM)   // 192

typedef __attribute__((ext_vector_type(16))) __bf16 bf16x16;
typedef __attribute__((ext_vector_type(8)))  float  f32x8;
typedef unsigned int v4u __attribute__((ext_vector_type(4)));
typedef int          v8i __attribute__((ext_vector_type(8)));
typedef int          v4i __attribute__((ext_vector_type(4)));

union FragAB {
    uint4   u[2];
    bf16x16 v;
};

#define HAVE_TDM (__has_builtin(__builtin_amdgcn_tensor_load_to_lds) && \
                  __has_builtin(__builtin_amdgcn_s_wait_tensorcnt))

__device__ __forceinline__ unsigned short f2bf(float f) {
    unsigned int u = __float_as_uint(f);
    unsigned int r = u + 0x7FFFu + ((u >> 16) & 1u);   // round-to-nearest-even
    return (unsigned short)(r >> 16);
}

#if HAVE_TDM
// Issue one TDM 2D tile load: `rows` rows x 32 bf16 cols, row stride 1024 elems,
// row-major into LDS at byte offset lds_off with 16B padding after each 64B row
// (pad_interval enc 3 = 16 DWORDs, pad_amount enc 3 = 4 DWORDs) -> LDS row stride 80B.
__device__ __forceinline__ void tdm_load_tile(unsigned long long gaddr, unsigned int lds_off,
                                              unsigned int rows, unsigned int tensor_rows) {
    v4u g0;
    g0.x = 1u;                                                   // count=1 (valid user D#)
    g0.y = lds_off;                                              // lds_addr [63:32]
    g0.z = (unsigned int)(gaddr & 0xFFFFFFFFu);                  // global_addr low
    g0.w = (unsigned int)((gaddr >> 32) & 0x01FFFFFFu) | (2u << 30);  // addr hi | type=2
    v8i g1;
    g1[0] = (int)((1u << 16) | (1u << 20) | (3u << 22) | (3u << 25)); // 2B elems, pad on, 16DW/4DW
    g1[1] = (int)((KDIM & 0xFFFFu) << 16);                       // tensor_dim0 = 1024 (lo16)
    g1[2] = (int)(((KDIM >> 16) & 0xFFFFu) | ((tensor_rows & 0xFFFFu) << 16));
    g1[3] = (int)(((tensor_rows >> 16) & 0xFFFFu) | (32u << 16));// tensor_dim1 hi | tile_dim0=32
    g1[4] = (int)(rows & 0xFFFFu);                               // tile_dim1 (tile_dim2=0)
    g1[5] = (int)KDIM;                                           // tensor_dim0_stride = 1024
    g1[6] = 0;
    g1[7] = 0;
    v4i z4 = {0, 0, 0, 0};
    v8i z8 = {0, 0, 0, 0, 0, 0, 0, 0};
    // amdgpu-toolchain (clang-23) 6-arg form: (g0, g1, g2, g3, g4, cpol)
    __builtin_amdgcn_tensor_load_to_lds(g0, g1, z4, z4, z8, 0);
}
#endif

// value (L,B,E) f32 -> inp_bf (B,L,E) bf16 ; 4 elems per thread
__global__ __launch_bounds__(256) void k_cvt_value(const float* __restrict__ value,
                                                   unsigned short* __restrict__ inp_bf) {
    size_t o = ((size_t)blockIdx.x * blockDim.x + threadIdx.x) * 4;   // out flat (b,l,e)
    int b = (int)(o >> 22);
    int l = (int)((o >> 10) & 4095);
    int e = (int)(o & 1023);
    const float4 f = *(const float4*)(value + ((size_t)(l * B_SZ + b) * E_DIM + e));
    ushort4 s;
    s.x = f2bf(f.x); s.y = f2bf(f.y); s.z = f2bf(f.z); s.w = f2bf(f.w);
    *(ushort4*)(inp_bf + o) = s;
}

// flat f32 -> bf16 (weights); 4 elems per thread
__global__ __launch_bounds__(256) void k_cvt_flat(const float* __restrict__ in,
                                                  unsigned short* __restrict__ out) {
    size_t i = (size_t)blockIdx.x * blockDim.x + threadIdx.x;
    const float4 f = ((const float4*)in)[i];
    ushort4 s;
    s.x = f2bf(f.x); s.y = f2bf(f.y); s.z = f2bf(f.z); s.w = f2bf(f.w);
    ((ushort4*)out)[i] = s;
}

// WMMA bf16 GEMM:  Out[M,N] = A[M,K] * W[N,K]^T, f32 accumulate.
// Block tile 128 x (NF*32); 8 wave32 waves in a 4x2 grid; wave tile 32 x (NF*16).
// TDM double-buffered tile staging (tensor_load_to_lds + s_wait_tensorcnt) when available.
// mode 0: outF[gm*N + gn] = acc                       (V = inp @ W_in^T)
// mode 1: attng[gm*192+gn] = acc * mask[gm*192+gn]    (gated attention, flat reshape view)
// mode 2: outF[(l*B + b)*1024 + gn] = acc             (final out with (B,L)->(L,B) transpose)
template <int NF>
__global__ __launch_bounds__(256) void k_gemm(const unsigned short* __restrict__ A,
                                              const unsigned short* __restrict__ W,
                                              float* __restrict__ outF,
                                              float* __restrict__ attng,
                                              const float* __restrict__ mask,
                                              int N, int mode) {
    constexpr int BN      = NF * 32;              // 64 or 128
    constexpr int AWORDS  = 128 * 40;             // A tile: 128 rows x 32, LDS stride 40
    constexpr int BWORDS  = BN * 40;
    constexpr int BUFW    = AWORDS + BWORDS;      // one double-buffer slot (ushorts)
    __shared__ unsigned short smem[2 * BUFW];

    const int tid  = threadIdx.x;
    const int bM   = blockIdx.y * 128;
    const int bN   = blockIdx.x * BN;
    const int wave = tid >> 5, lane = tid & 31;
    const int wm   = wave >> 1, wn = wave & 1;    // 4x2 wave grid
    const int lhalf = lane >> 4, lidx = lane & 15;
    const int kb    = lhalf * 8;                  // ISA 16-bit A/B frag K-base per lane half

    f32x8 c[2][NF];
#pragma unroll
    for (int mf = 0; mf < 2; ++mf)
#pragma unroll
        for (int nf = 0; nf < NF; ++nf)
#pragma unroll
            for (int r = 0; r < 8; ++r) c[mf][nf][r] = 0.0f;

#if HAVE_TDM
    // ---- TDM double-buffered pipeline: wave 0 drives the Tensor Data Mover ----
    if (wave == 0) {
        tdm_load_tile((unsigned long long)(const void*)(A + (size_t)bM * KDIM),
                      0u, 128u, (unsigned int)MROWS);
        tdm_load_tile((unsigned long long)(const void*)(W + (size_t)bN * KDIM),
                      (unsigned int)(AWORDS * 2), (unsigned int)BN, (unsigned int)N);
    }
    int buf = 0;
    for (int kt = 0; kt < KDIM; kt += 32, buf ^= 1) {
        const unsigned short* Abase = smem + buf * BUFW;
        const unsigned short* Bbase = Abase + AWORDS;
        if (wave == 0) {
            if (kt + 32 < KDIM) {
                const unsigned int nb = (unsigned int)((buf ^ 1) * BUFW * 2);
                tdm_load_tile((unsigned long long)(const void*)(A + (size_t)bM * KDIM + kt + 32),
                              nb, 128u, (unsigned int)MROWS);
                tdm_load_tile((unsigned long long)(const void*)(W + (size_t)bN * KDIM + kt + 32),
                              nb + (unsigned int)(AWORDS * 2), (unsigned int)BN, (unsigned int)N);
                __builtin_amdgcn_s_wait_tensorcnt(2);   // current pair done, next pair in flight
            } else {
                __builtin_amdgcn_s_wait_tensorcnt(0);
            }
        }
        __syncthreads();   // current buffer visible to all waves
#else
    // ---- fallback: per-thread global->LDS staging (single buffer) ----
    for (int kt = 0; kt < KDIM; kt += 32) {
        const unsigned short* Abase = smem;
        const unsigned short* Bbase = smem + AWORDS;
#pragma unroll
        for (int q = tid; q < 128 * 4; q += 256) {
            const int row = q >> 2, seg = q & 3;
            *(uint4*)(&smem[row * 40 + seg * 8]) =
                *(const uint4*)(A + (size_t)(bM + row) * KDIM + kt + seg * 8);
        }
#pragma unroll
        for (int q = tid; q < BN * 4; q += 256) {
            const int row = q >> 2, seg = q & 3;
            *(uint4*)(&smem[AWORDS + row * 40 + seg * 8]) =
                *(const uint4*)(W + (size_t)(bN + row) * KDIM + kt + seg * 8);
        }
        if (kt + 32 < KDIM) {
            __builtin_prefetch(A + (size_t)(bM + (tid >> 1)) * KDIM + kt + 32, 0, 1);
            __builtin_prefetch(W + (size_t)(bN + (tid >> 2)) * KDIM + kt + 32, 0, 1);
        }
        __syncthreads();
#endif

        // ---- fragments per ISA 16-bit layouts; 2 x NF WMMA per K-chunk ----
        FragAB a[2], b[NF];
#pragma unroll
        for (int mf = 0; mf < 2; ++mf) {
            const int m = wm * 32 + mf * 16 + lidx;
            a[mf].u[0] = *(const uint4*)(&Abase[m * 40 + kb]);
            a[mf].u[1] = *(const uint4*)(&Abase[m * 40 + kb + 16]);
        }
#pragma unroll
        for (int nf = 0; nf < NF; ++nf) {
            const int n = wn * (16 * NF) + nf * 16 + lidx;
            b[nf].u[0] = *(const uint4*)(&Bbase[n * 40 + kb]);
            b[nf].u[1] = *(const uint4*)(&Bbase[n * 40 + kb + 16]);
        }
#pragma unroll
        for (int mf = 0; mf < 2; ++mf)
#pragma unroll
            for (int nf = 0; nf < NF; ++nf)
                c[mf][nf] = __builtin_amdgcn_wmma_f32_16x16x32_bf16(
                    /*neg_a=*/false, a[mf].v, /*neg_b=*/false, b[nf].v,
                    /*c_mod=*/(short)0, c[mf][nf],
                    /*reuse_a=*/false, /*reuse_b=*/false);
        __syncthreads();   // safe to overwrite staged tile
    }

    // ---- epilogue ----
#pragma unroll
    for (int mf = 0; mf < 2; ++mf)
#pragma unroll
        for (int nf = 0; nf < NF; ++nf)
#pragma unroll
            for (int r = 0; r < 8; ++r) {
                const int gm = bM + wm * 32 + mf * 16 + r + lhalf * 8;
                const int gn = bN + wn * (16 * NF) + nf * 16 + lidx;
                const float v = c[mf][nf][r];
                if (mode == 0) {
                    outF[(size_t)gm * N + gn] = v;
                } else if (mode == 1) {
                    const size_t o = (size_t)gm * NATT + gn;   // flat reshape view
                    attng[o] = v * mask[o];
                } else {
                    const int bb = gm >> 12, ll = gm & 4095;
                    outF[((size_t)(ll * B_SZ + bb)) * E_DIM + gn] = v;
                }
            }
}

// 12-level shift-gate-accumulate scan. Flat buffers viewed as (64, 4096, 64) / (64, 4096, 12)
// (identical row-major scramble on both sides, so no unscrambling needed).
// One block owns (bh, 2 d-lanes) over all L: 4096x2 f32 = 32KB LDS, register-staged updates.
__global__ __launch_bounds__(256) void k_scan(const float* __restrict__ Vf,
                                              const float* __restrict__ attng,
                                              unsigned short* __restrict__ Vbf) {
    const int bh = blockIdx.x;          // 0..63
    const int d0 = blockIdx.y * 2;      // 0..62
    __shared__ float sV[L_SEQ * 2];
    const int col = threadIdx.x & 1;
    const int ls  = threadIdx.x >> 1;   // 0..127

#pragma unroll 4
    for (int j = 0; j < 32; ++j) {
        const int l = ls + j * 128;
        sV[l * 2 + col] = Vf[((size_t)bh * L_SEQ + l) * HD + d0 + col];
    }
    __syncthreads();

    float nv[32];
    for (int i = 0; i < LVLS; ++i) {
        const int shift = 1 << i;
#pragma unroll 4
        for (int j = 0; j < 32; ++j) {
            const int l  = ls + j * 128;
            const int lp = (l - shift) & (L_SEQ - 1);
            const float g = attng[((size_t)bh * L_SEQ + l) * LVLS + i];
            nv[j] = sV[l * 2 + col] + g * sV[lp * 2 + col];
        }
        __syncthreads();
#pragma unroll 4
        for (int j = 0; j < 32; ++j) sV[(ls + j * 128) * 2 + col] = nv[j];
        __syncthreads();
    }

#pragma unroll 4
    for (int j = 0; j < 32; ++j) {
        const int l = ls + j * 128;
        Vbf[((size_t)bh * L_SEQ + l) * HD + d0 + col] = f2bf(sV[l * 2 + col]);
    }
}

extern "C" void kernel_launch(void* const* d_in, const int* in_sizes, int n_in,
                              void* d_out, int out_size, void* d_ws, size_t ws_size,
                              hipStream_t stream) {
    (void)in_sizes; (void)n_in; (void)out_size; (void)ws_size;
    const float* value = (const float*)d_in[0];
    const float* mask  = (const float*)d_in[1];
    const float* W_in  = (const float*)d_in[2];
    const float* W_att = (const float*)d_in[3];
    const float* W_out = (const float*)d_in[4];
    float* out = (float*)d_out;

    char* ws = (char*)d_ws;
    size_t off = 0;
    auto alloc = [&](size_t bytes) {
        void* p = ws + off;
        off = (off + bytes + 255) & ~(size_t)255;
        return p;
    };
    unsigned short* inp_bf  = (unsigned short*)alloc((size_t)MROWS * E_DIM * 2);   // 32 MB
    unsigned short* win_bf  = (unsigned short*)alloc((size_t)E_DIM * E_DIM * 2);   //  2 MB
    unsigned short* watt_bf = (unsigned short*)alloc((size_t)NATT * E_DIM * 2);
    unsigned short* wout_bf = (unsigned short*)alloc((size_t)E_DIM * E_DIM * 2);
    float*          Vf      = (float*)alloc((size_t)MROWS * E_DIM * 4);            // 64 MB
    float*          attng   = (float*)alloc((size_t)MROWS * NATT * 4);             // 12.6 MB
    unsigned short* Vbf     = (unsigned short*)alloc((size_t)MROWS * E_DIM * 2);   // 32 MB

    // 1) precision/layout conversion (f32 -> bf16, value transposed to (B,L,E))
    k_cvt_value<<<(MROWS * E_DIM) / 4 / 256, 256, 0, stream>>>(value, inp_bf);
    k_cvt_flat<<<(E_DIM * E_DIM) / 4 / 256, 256, 0, stream>>>(W_in, win_bf);
    k_cvt_flat<<<(NATT * E_DIM) / 4 / 256, 256, 0, stream>>>(W_att, watt_bf);
    k_cvt_flat<<<(E_DIM * E_DIM) / 4 / 256, 256, 0, stream>>>(W_out, wout_bf);

    // 2) V = inp @ W_in^T   (16384 x 1024), 128x128 block tiles
    k_gemm<4><<<dim3(E_DIM / 128, MROWS / 128), 256, 0, stream>>>(
        inp_bf, win_bf, Vf, nullptr, nullptr, E_DIM, 0);

    // 3) attention = (inp @ W_att^T) * mask   (16384 x 192), 128x64 block tiles
    k_gemm<2><<<dim3(NATT / 64, MROWS / 128), 256, 0, stream>>>(
        inp_bf, watt_bf, nullptr, attng, mask, NATT, 1);

    // 4) 12-level scan, emits bf16 V for the final GEMM
    k_scan<<<dim3(64, 32), 256, 0, stream>>>(Vf, attng, Vbf);

    // 5) out = V @ W_out^T, fused (B,L,E) -> (L,B,E) transpose on store
    k_gemm<4><<<dim3(E_DIM / 128, MROWS / 128), 256, 0, stream>>>(
        Vbf, wout_bf, out, nullptr, nullptr, E_DIM, 2);
}
